// MultiHeadAttentionBlock_41059887349946
// MI455X (gfx1250) — matrix operand order repaired and checked by reading, not compiled, not run on hardware
//
#include <hip/hip_runtime.h>
#include <hip/hip_bf16.h>
#include <cstddef>

typedef __attribute__((ext_vector_type(16))) _Float16 v16h;
typedef __attribute__((ext_vector_type(8)))  float    v8f;

#define WMMA_F32_F16(A, B, C) \
    __builtin_amdgcn_wmma_f32_16x16x32_f16(false, (A), false, (B), (short)0, (C), false, false)

// 32-bit LDS byte address (addrspace(3) offset) of a generic pointer to LDS
__device__ __forceinline__ unsigned lds_addr(const void* p) {
    return (unsigned)(size_t)(__attribute__((address_space(3))) const void*)p;
}

// async copy of 16 bytes global -> LDS (tracked by ASYNCcnt)
__device__ __forceinline__ void async_copy_b128(unsigned lds_dst, const void* gsrc) {
    asm volatile("global_load_async_to_lds_b128 %0, %1, off"
                 :: "v"(lds_dst), "v"(gsrc) : "memory");
}
__device__ __forceinline__ void wait_asynccnt0() {
    asm volatile("s_wait_asynccnt 0" ::: "memory");
}

// ---------------------------------------------------------------------------
// f32 -> f16 conversion (x and weights), elementwise
// ---------------------------------------------------------------------------
__global__ void __launch_bounds__(256)
f32_to_f16_kernel(const float* __restrict__ in, _Float16* __restrict__ out, int n) {
    int i = blockIdx.x * 256 + threadIdx.x;
    if (i < n) out[i] = (_Float16)in[i];
}

// ---------------------------------------------------------------------------
// Tiled WMMA GEMM: C[M,N] = A[M,K] (f16, row-major) * B[K,N] (f16, row-major)
// Block tile 128x64, K-chunk 32, 8 waves (4x2), each wave 32x32 (2x2 WMMA tiles)
// Double-buffered LDS: A tile via async-copy engine (ASYNCcnt), B tile via
// register pipeline (needs a transpose into LDS, which async copies can't do).
// OUT_F32=false: Ch = (f16)(acc + bias)        (QKV projections)
// OUT_F32=true : Cf = acc + bias + res         (output projection + residual)
// ---------------------------------------------------------------------------
template <bool OUT_F32>
__global__ void __launch_bounds__(256)
gemm_wmma_kernel(const _Float16* __restrict__ A,
                 const _Float16* __restrict__ B,
                 const float* __restrict__ bias,
                 const float* __restrict__ res,
                 _Float16* __restrict__ Ch,
                 float* __restrict__ Cf,
                 int M, int N, int K)
{
    constexpr int BM = 128, BN = 64, BK = 32;
    __shared__ __align__(16) _Float16 As[2][BM][BK + 8];   // [buf][m][k]
    __shared__ __align__(16) _Float16 Bs[2][BN][BK + 8];   // [buf][n][k] (transposed)

    const int tid  = threadIdx.x;
    const int lane = tid & 31;
    const int wave = tid >> 5;
    const int wm   = wave & 3;                // wave row (4 along M)
    const int wn   = wave >> 2;               // wave col (2 along N)
    const int bm0  = blockIdx.x * BM;
    const int bn0  = blockIdx.y * BN;
    const int lhi  = lane >> 4;               // 0 or 1 (wave half)
    const int llo  = lane & 15;

    // B staging geometry: 8 contiguous halves along n per thread
    const int bk_ = (tid * 8) >> 6;           // k row within tile (0..31)
    const int bn_ = (tid * 8) & 63;           // n start within tile

    // A tile: 4096 halves = 512 x 16B chunks, 2 async b128 copies per thread
    auto issueA = [&](int buf, int k0) {
        #pragma unroll
        for (int i = 0; i < 2; ++i) {
            int id = tid * 2 + i;
            int r = id >> 2, c = (id & 3) << 3;
            async_copy_b128(lds_addr(&As[buf][r][c]),
                            A + (size_t)(bm0 + r) * K + k0 + c);
        }
    };
    _Float16 rb[8];
    auto loadBregs = [&](int k0) {
        const _Float16* src = B + (size_t)(k0 + bk_) * N + bn0 + bn_;
        #pragma unroll
        for (int i = 0; i < 8; ++i) rb[i] = src[i];
    };
    auto storeBregs = [&](int buf) {
        #pragma unroll
        for (int i = 0; i < 8; ++i) Bs[buf][bn_ + i][bk_] = rb[i];
    };

    v8f acc[2][2] = {};

    // prologue: stage tile 0
    issueA(0, 0);
    loadBregs(0);
    storeBregs(0);
    wait_asynccnt0();
    __syncthreads();

    const int ntiles = K / BK;
    int cur = 0;
    for (int t = 0; t < ntiles; ++t) {
        const int nxt = cur ^ 1;
        const bool more = (t + 1) < ntiles;
        if (more) {                            // tile t+1 in flight under compute
            issueA(nxt, (t + 1) * BK);
            loadBregs((t + 1) * BK);
        }

        v16h af[2], bf[2];
        #pragma unroll
        for (int u = 0; u < 2; ++u) {          // A fragments (16x32): row=llo, K=lhi*16+j
            const _Float16* p = &As[cur][wm * 32 + u * 16 + llo][lhi * 16];
            #pragma unroll
            for (int j = 0; j < 8; ++j) { af[u][j] = p[j]; af[u][8 + j] = p[8 + j]; }
        }
        #pragma unroll
        for (int u = 0; u < 2; ++u) {          // B fragments (32x16): col=llo, K=lhi*16+j
            const _Float16* p = &Bs[cur][wn * 32 + u * 16 + llo][lhi * 16];
            #pragma unroll
            for (int j = 0; j < 16; ++j) bf[u][j] = p[j];
        }
        #pragma unroll
        for (int i = 0; i < 2; ++i)
            #pragma unroll
            for (int j = 0; j < 2; ++j)
                acc[i][j] = WMMA_F32_F16(af[i], bf[j], acc[i][j]);

        if (more) {
            storeBregs(nxt);
            wait_asynccnt0();                  // tile t+1 A data landed
        }
        __syncthreads();
        cur = nxt;
    }

    // epilogue (C layout: row = v + 8*lhi, col = llo)
    #pragma unroll
    for (int i = 0; i < 2; ++i)
        #pragma unroll
        for (int j = 0; j < 2; ++j) {
            int col = bn0 + wn * 32 + j * 16 + llo;
            #pragma unroll
            for (int v = 0; v < 8; ++v) {
                int row = bm0 + wm * 32 + i * 16 + v + (lhi << 3);
                float val = acc[i][j][v] + bias[col];
                if constexpr (OUT_F32)
                    Cf[(size_t)row * N + col] = val + res[(size_t)row * N + col];
                else
                    Ch[(size_t)row * N + col] = (_Float16)val;
            }
        }
}

// ---------------------------------------------------------------------------
// Flash attention, causal. One wave owns a 16-query tile of one (b,h).
// Key blocks of 32 -> P(16x32) is exactly one WMMA A fragment.
// Q/K/V/Ctx are flat (B,H,S,64) f16 (== the reference's reshape for free).
// ---------------------------------------------------------------------------
__global__ void __launch_bounds__(256)
flash_attn_kernel(const _Float16* __restrict__ Q,
                  const _Float16* __restrict__ Km,
                  const _Float16* __restrict__ Vm,
                  _Float16* __restrict__ Ctx)
{
    constexpr int S = 2048;
    __shared__ _Float16 Pbuf[8][16][34];      // per-wave P slab (C-layout -> A-layout)

    const int lane = threadIdx.x & 31;
    const int wave = threadIdx.x >> 5;
    const int lhi  = lane >> 4;
    const int llo  = lane & 15;
    const int gw   = blockIdx.x * 8 + wave;   // global wave id
    const int qt   = gw & 127;                // S/16 = 128 query tiles
    const int bh   = gw >> 7;                 // 0..31 (B*H)
    const int q0   = qt * 16;
    const size_t base = (size_t)bh * S * 64;

    // Q fragments: A-layout, rows q0+llo, K = d
    v16h qf[2];
    {
        const _Float16* qrow = Q + base + (size_t)(q0 + llo) * 64;
        #pragma unroll
        for (int c = 0; c < 2; ++c) {
            const _Float16* p = qrow + c * 32 + lhi * 16;
            #pragma unroll
            for (int j = 0; j < 8; ++j) { qf[c][j] = p[j]; qf[c][8 + j] = p[8 + j]; }
        }
    }

    v8f o[4] = {};
    float mrow[8], lrow[8];
    #pragma unroll
    for (int v = 0; v < 8; ++v) { mrow[v] = -3.0e38f; lrow[v] = 0.0f; }

    for (int kb = 0; kb <= q0 + 15; kb += 32) {
        // prefetch next key block (speculative, L2-bound)
        if (kb + 32 < S) {
            __builtin_prefetch(Km + base + (size_t)(kb + 32 + llo) * 64, 0, 1);
            __builtin_prefetch(Vm + base + (size_t)(kb + 32 + llo) * 64, 0, 1);
        }

        // scores: two 16x16 tiles, S = Q * K^T / sqrt(dk)
        v8f s[2];
        #pragma unroll
        for (int t = 0; t < 2; ++t) {
            v8f z = {};
            #pragma unroll
            for (int c = 0; c < 2; ++c) {
                v16h bf;  // B[d][n] = K[kb + t*16 + n][c*32 + d]; col = llo, d = lhi*16 + j
                const _Float16* p =
                    Km + base + (size_t)(kb + t * 16 + llo) * 64 + c * 32 + lhi * 16;
                #pragma unroll
                for (int j = 0; j < 16; ++j) bf[j] = p[j];
                z = WMMA_F32_F16(qf[c], bf, z);
            }
            s[t] = z;
        }

        const bool needMask = (kb + 31) > q0;  // block not fully below diagonal
        float pv[2][8];
        #pragma unroll
        for (int v = 0; v < 8; ++v) {
            const int qrow_ = q0 + v + (lhi << 3);
            float x0 = s[0][v] * 0.125f;
            float x1 = s[1][v] * 0.125f;
            if (needMask) {
                if (kb + llo      > qrow_) x0 = -1.0e30f;
                if (kb + 16 + llo > qrow_) x1 = -1.0e30f;
            }
            // row max across the 16 lanes of this half-wave (N dimension)
            float rm = fmaxf(x0, x1);
            #pragma unroll
            for (int off = 8; off >= 1; off >>= 1) rm = fmaxf(rm, __shfl_xor(rm, off, 32));
            float mnew = fmaxf(mrow[v], rm);
            float corr = __expf(mrow[v] - mnew);
            float p0 = __expf(x0 - mnew);
            float p1 = __expf(x1 - mnew);
            float rs = p0 + p1;
            #pragma unroll
            for (int off = 8; off >= 1; off >>= 1) rs += __shfl_xor(rs, off, 32);
            lrow[v] = lrow[v] * corr + rs;
            mrow[v] = mnew;
            #pragma unroll
            for (int jj = 0; jj < 4; ++jj) o[jj][v] *= corr;
            pv[0][v] = p0; pv[1][v] = p1;
        }

        // C-layout -> A-layout via per-wave LDS slab (no block barrier: waves
        // have different trip counts; use explicit dscnt wait instead)
        #pragma unroll
        for (int v = 0; v < 8; ++v) {
            int r = v + (lhi << 3);
            Pbuf[wave][r][llo]      = (_Float16)pv[0][v];
            Pbuf[wave][r][16 + llo] = (_Float16)pv[1][v];
        }
        asm volatile("s_wait_dscnt 0" ::: "memory");
        v16h pa;
        {
            const _Float16* p = &Pbuf[wave][llo][lhi * 16];
            #pragma unroll
            for (int j = 0; j < 8; ++j) { pa[j] = p[j]; pa[8 + j] = p[8 + j]; }
        }

        // O += P(16x32) * V(32x64): 4 WMMAs over the DV dimension
        #pragma unroll
        for (int jj = 0; jj < 4; ++jj) {
            v16h vb;  // B[k][n] = V[kb + k][jj*16 + n]; col = llo, k = lhi*16 + j
            #pragma unroll
            for (int j = 0; j < 16; ++j)
                vb[j] = Vm[base + (size_t)(kb + (lhi << 4) + j) * 64 + jj * 16 + llo];
            o[jj] = WMMA_F32_F16(pa, vb, o[jj]);
        }
    }

    // normalize and write context (flat (B,H,S,64) == (B,S,H*64) for the next GEMM)
    #pragma unroll
    for (int jj = 0; jj < 4; ++jj) {
        int col = jj * 16 + llo;
        #pragma unroll
        for (int v = 0; v < 8; ++v) {
            int r = q0 + v + (lhi << 3);
            Ctx[base + (size_t)r * 64 + col] = (_Float16)(o[jj][v] / lrow[v]);
        }
    }
}

// ---------------------------------------------------------------------------
// Row-wise LayerNorm over D=1024: one block (256 thr) per row
// ---------------------------------------------------------------------------
__global__ void __launch_bounds__(256)
layernorm_kernel(const float* __restrict__ inp, const float* __restrict__ gamma,
                 const float* __restrict__ beta, float* __restrict__ out)
{
    __shared__ float red[2][8];
    const int row = blockIdx.x;
    const int tid = threadIdx.x;
    const float* x = inp + (size_t)row * 1024;

    float vals[4];
    float s = 0.0f, ss = 0.0f;
    #pragma unroll
    for (int i = 0; i < 4; ++i) {
        float v = x[tid + 256 * i];
        vals[i] = v; s += v; ss += v * v;
    }
    #pragma unroll
    for (int off = 16; off >= 1; off >>= 1) {
        s  += __shfl_xor(s,  off, 32);
        ss += __shfl_xor(ss, off, 32);
    }
    if ((tid & 31) == 0) { red[0][tid >> 5] = s; red[1][tid >> 5] = ss; }
    __syncthreads();
    float ts = 0.0f, tss = 0.0f;
    #pragma unroll
    for (int i = 0; i < 8; ++i) { ts += red[0][i]; tss += red[1][i]; }
    float mean = ts * (1.0f / 1024.0f);
    float var  = tss * (1.0f / 1024.0f) - mean * mean;
    float rstd = rsqrtf(var + 1e-5f);
    #pragma unroll
    for (int i = 0; i < 4; ++i) {
        int c = tid + 256 * i;
        out[(size_t)row * 1024 + c] = (vals[i] - mean) * rstd * gamma[c] + beta[c];
    }
}

// ---------------------------------------------------------------------------
// Host orchestration
// ---------------------------------------------------------------------------
extern "C" void kernel_launch(void* const* d_in, const int* in_sizes, int n_in,
                              void* d_out, int out_size, void* d_ws, size_t ws_size,
                              hipStream_t stream)
{
    (void)in_sizes; (void)n_in; (void)out_size; (void)ws_size;
    const float* x     = (const float*)d_in[0];
    const float* Wk    = (const float*)d_in[1];
    const float* bk    = (const float*)d_in[2];
    const float* Wq    = (const float*)d_in[3];
    const float* bq    = (const float*)d_in[4];
    const float* Wv    = (const float*)d_in[5];
    const float* bv    = (const float*)d_in[6];
    const float* Wo    = (const float*)d_in[7];
    const float* bo    = (const float*)d_in[8];
    const float* gamma = (const float*)d_in[9];
    const float* beta  = (const float*)d_in[10];
    float* out = (float*)d_out;

    const int M = 2 * 2048;          // B*S rows
    const int D = 1024;
    const size_t MD = (size_t)M * D; // 4M elements
    const size_t DD = (size_t)D * D;

    char* ws = (char*)d_ws;
    _Float16* xh  = (_Float16*)ws; ws += MD * 2;
    _Float16* wqh = (_Float16*)ws; ws += DD * 2;
    _Float16* wkh = (_Float16*)ws; ws += DD * 2;
    _Float16* wvh = (_Float16*)ws; ws += DD * 2;
    _Float16* woh = (_Float16*)ws; ws += DD * 2;
    _Float16* Qh  = (_Float16*)ws; ws += MD * 2;
    _Float16* Kh  = (_Float16*)ws; ws += MD * 2;
    _Float16* Vh  = (_Float16*)ws; ws += MD * 2;
    _Float16* Ctx = (_Float16*)ws; ws += MD * 2;
    float*    pre = (float*)ws;    ws += MD * 4;

    f32_to_f16_kernel<<<(int)((MD + 255) / 256), 256, 0, stream>>>(x,  xh,  (int)MD);
    f32_to_f16_kernel<<<(int)((DD + 255) / 256), 256, 0, stream>>>(Wq, wqh, (int)DD);
    f32_to_f16_kernel<<<(int)((DD + 255) / 256), 256, 0, stream>>>(Wk, wkh, (int)DD);
    f32_to_f16_kernel<<<(int)((DD + 255) / 256), 256, 0, stream>>>(Wv, wvh, (int)DD);
    f32_to_f16_kernel<<<(int)((DD + 255) / 256), 256, 0, stream>>>(Wo, woh, (int)DD);

    dim3 g(M / 128, D / 64);
    gemm_wmma_kernel<false><<<g, 256, 0, stream>>>(xh, wqh, bq, nullptr, Qh, nullptr, M, D, D);
    gemm_wmma_kernel<false><<<g, 256, 0, stream>>>(xh, wkh, bk, nullptr, Kh, nullptr, M, D, D);
    gemm_wmma_kernel<false><<<g, 256, 0, stream>>>(xh, wvh, bv, nullptr, Vh, nullptr, M, D, D);

    flash_attn_kernel<<<512, 256, 0, stream>>>(Qh, Kh, Vh, Ctx);

    gemm_wmma_kernel<true><<<g, 256, 0, stream>>>(Ctx, woh, bo, x, nullptr, pre, M, D, D);

    layernorm_kernel<<<M, 256, 0, stream>>>(pre, gamma, beta, out);
}
